// LinformerAttentionMethod_32469952757997
// MI455X (gfx1250) — matrix operand order repaired and checked by reading, not compiled, not run on hardware
//
#include <hip/hip_runtime.h>
#include <hip/hip_bf16.h>
#include <math.h>

// ---------------- problem constants ----------------
#define BATCH 8
#define HEADS 16
#define SEQ   4096
#define DH    64      // head dim
#define KP    256     // linformer projected dim
#define BH    (BATCH*HEADS)

typedef __bf16 bf16_t;
typedef __attribute__((ext_vector_type(16))) __bf16 v16bf;
typedef __attribute__((ext_vector_type(8)))  __bf16 v8bf;
typedef __attribute__((ext_vector_type(4)))  __bf16 v4bf;
typedef __attribute__((ext_vector_type(8)))  float  v8f;
typedef __attribute__((ext_vector_type(4)))  float  v4f;

// Build a 16-element bf16 A/B fragment from two contiguous 16-byte LDS runs.
__device__ __forceinline__ v16bf frag2(const bf16_t* p0, const bf16_t* p1) {
    v8bf a = *(const v8bf*)p0;
    v8bf b = *(const v8bf*)p1;
    return __builtin_shufflevector(a, b, 0,1,2,3,4,5,6,7,8,9,10,11,12,13,14,15);
}

__device__ __forceinline__ v8f wmma_bf16(v16bf a, v16bf b, v8f c) {
    return __builtin_amdgcn_wmma_f32_16x16x32_bf16(
        /*neg_a=*/false, a, /*neg_b=*/false, b,
        /*c_mod=*/(short)0, c, /*reuse_a=*/false, /*reuse_b=*/false);
}

// =====================================================================
// Kernel 1: per-(b,h) projection GEMMs with split-K over the sequence.
//   Kp[kk,d] = sum_n proj_k[n,kk] * k[b,h,n,d] * mask[b,n]   (and same for V)
// M = 256 (kk), N = 64 (d), Kdim = 4096 (n) split 8 ways, bf16 WMMA,
// fp32 atomic accumulation into workspace.
// =====================================================================
__global__ __launch_bounds__(256)
void linf_proj_kernel(const float* __restrict__ kmat, const float* __restrict__ vmat,
                      const float* __restrict__ mask,
                      const float* __restrict__ pk,   const float* __restrict__ pv,
                      float* __restrict__ wsK, float* __restrict__ wsV)
{
    // A tiles (proj^T), stored row-major by M (proj column), 32 n's per row
    __shared__ bf16_t Apk[KP * 32];   // 16 KB
    __shared__ bf16_t Apv[KP * 32];   // 16 KB
    // B tiles (k/v), stored transposed: [d][n-offset]
    __shared__ bf16_t Bk[DH * 32];    // 4 KB
    __shared__ bf16_t Bv[DH * 32];    // 4 KB

    const int bh    = blockIdx.x;            // 0..127
    const int b     = bh >> 4;
    const int split = blockIdx.y;            // 0..7
    const int tid   = threadIdx.x;
    const int wave  = tid >> 5;
    const int lane  = tid & 31;
    const int l15   = lane & 15;
    const int half  = lane >> 4;

    const float* kbase = kmat + (size_t)bh * SEQ * DH;
    const float* vbase = vmat + (size_t)bh * SEQ * DH;
    const float* mbase = mask + (size_t)b * SEQ;

    v8f accK[2][4];
    v8f accV[2][4];
#pragma unroll
    for (int mi = 0; mi < 2; ++mi)
#pragma unroll
        for (int nt = 0; nt < 4; ++nt) {
            accK[mi][nt] = (v8f)(0.0f);
            accV[mi][nt] = (v8f)(0.0f);
        }

    for (int ni = 0; ni < 16; ++ni) {
        const int n0 = split * 512 + ni * 32;
        __syncthreads();   // previous iteration's LDS reads complete

        // ---- stage proj_k / proj_v columns (transpose 32x256 -> [m][c]) ----
        {
            v8bf ck[4], cv[4];
#pragma unroll
            for (int c = 0; c < 32; ++c) {
                float fk = pk[(size_t)(n0 + c) * KP + tid];
                float fv = pv[(size_t)(n0 + c) * KP + tid];
                ck[c >> 3][c & 7] = (bf16_t)fk;
                cv[c >> 3][c & 7] = (bf16_t)fv;
            }
#pragma unroll
            for (int q8 = 0; q8 < 4; ++q8) {
                *(v8bf*)&Apk[tid * 32 + q8 * 8] = ck[q8];
                *(v8bf*)&Apv[tid * 32 + q8 * 8] = cv[q8];
            }
        }
        // ---- stage k / v tiles (masked, transposed to [d][c]) ----
        {
            const int d  = tid & 63;
            const int c0 = (tid >> 6) * 8;
            v8bf bk, bv;
#pragma unroll
            for (int i = 0; i < 8; ++i) {
                const int c = c0 + i;
                const float mv = mbase[n0 + c];
                bk[i] = (bf16_t)(kbase[(size_t)(n0 + c) * DH + d] * mv);
                bv[i] = (bf16_t)(vbase[(size_t)(n0 + c) * DH + d] * mv);
            }
            *(v8bf*)&Bk[d * 32 + c0] = bk;
            *(v8bf*)&Bv[d * 32 + c0] = bv;
        }
        __syncthreads();

        // ---- compute: wave handles M-tiles {2w, 2w+1} x 4 N-tiles x {K,V} ----
        v16bf aK[2], aV[2];
#pragma unroll
        for (int mi = 0; mi < 2; ++mi) {
            const int m = (wave * 2 + mi) * 16 + l15;
            aK[mi] = frag2(&Apk[m * 32 + half * 8], &Apk[m * 32 + 16 + half * 8]);
            aV[mi] = frag2(&Apv[m * 32 + half * 8], &Apv[m * 32 + 16 + half * 8]);
        }
#pragma unroll
        for (int nt = 0; nt < 4; ++nt) {
            const int d = nt * 16 + l15;
            v16bf bK = frag2(&Bk[d * 32 + half * 8], &Bk[d * 32 + 16 + half * 8]);
            v16bf bV = frag2(&Bv[d * 32 + half * 8], &Bv[d * 32 + 16 + half * 8]);
#pragma unroll
            for (int mi = 0; mi < 2; ++mi) {
                accK[mi][nt] = wmma_bf16(aK[mi], bK, accK[mi][nt]);
                accV[mi][nt] = wmma_bf16(aV[mi], bV, accV[mi][nt]);
            }
        }
    }

    // ---- fp32 atomic accumulation of split-K partials ----
    float* Kdst = wsK + (size_t)bh * KP * DH;
    float* Vdst = wsV + (size_t)bh * KP * DH;
#pragma unroll
    for (int mi = 0; mi < 2; ++mi)
#pragma unroll
        for (int nt = 0; nt < 4; ++nt)
#pragma unroll
            for (int j = 0; j < 8; ++j) {
                const int row = (wave * 2 + mi) * 16 + j + 8 * half;  // kk
                const int col = nt * 16 + l15;                        // d
                unsafeAtomicAdd(&Kdst[row * DH + col], accK[mi][nt][j]);
                unsafeAtomicAdd(&Vdst[row * DH + col], accV[mi][nt][j]);
            }
}

// =====================================================================
// Kernel 2: scores = q @ Kp^T / 8, softmax, out = probs @ Vp.
// Block = 8 waves x 16 query rows = 128 rows. Softmax fully in registers
// using the WMMA C-layout + __shfl_xor over 16-lane half groups.
// =====================================================================
__global__ __launch_bounds__(256)
void linf_attn_kernel(const float* __restrict__ q,
                      const float* __restrict__ wsK, const float* __restrict__ wsV,
                      float* __restrict__ out, float* __restrict__ probs_out)
{
    __shared__ bf16_t Kpb[KP * DH];       // [kk][d]  row-major, 32 KB
    __shared__ bf16_t Vptb[DH * KP];      // [d][kk]  transposed, 32 KB
    __shared__ bf16_t P[8][16 * KP];      // per-wave probs (A-layout rows), 64 KB

    const int bh   = blockIdx.x;          // 0..127
    const int qblk = blockIdx.y;          // 0..31
    const int tid  = threadIdx.x;
    const int wave = tid >> 5;
    const int lane = tid & 31;
    const int l15  = lane & 15;
    const int half = lane >> 4;

    const float* Ksrc = wsK + (size_t)bh * KP * DH;
    const float* Vsrc = wsV + (size_t)bh * KP * DH;

    // ---- stage Kp (row-major) and Vp (transposed) as bf16 ----
#pragma unroll
    for (int j = 0; j < 16; ++j) {
        const int e4 = tid + j * 256;             // float4 index, 4096 total
        v4f xk = *(const v4f*)&Ksrc[e4 * 4];
        v4bf yk;
#pragma unroll
        for (int i = 0; i < 4; ++i) yk[i] = (bf16_t)xk[i];
        *(v4bf*)&Kpb[e4 * 4] = yk;

        v4f xv = *(const v4f*)&Vsrc[e4 * 4];
        const int kk = (e4 * 4) >> 6;             // row of Vp
        const int d0 = (e4 * 4) & 63;             // col of Vp
#pragma unroll
        for (int i = 0; i < 4; ++i) Vptb[(d0 + i) * KP + kk] = (bf16_t)xv[i];
    }
    __syncthreads();

    // ---- q A-fragments straight from global (bf16 convert in regs) ----
    const int m0 = qblk * 128 + wave * 16;
    const float* qrow = q + ((size_t)bh * SEQ + m0 + l15) * DH;
    v16bf qf[2];
#pragma unroll
    for (int ki = 0; ki < 2; ++ki) {
        v4f a0 = *(const v4f*)&qrow[ki * 32 + half * 8];
        v4f a1 = *(const v4f*)&qrow[ki * 32 + half * 8 + 4];
        v4f a2 = *(const v4f*)&qrow[ki * 32 + 16 + half * 8];
        v4f a3 = *(const v4f*)&qrow[ki * 32 + 16 + half * 8 + 4];
        v16bf f;
#pragma unroll
        for (int i = 0; i < 4; ++i) {
            f[i]      = (bf16_t)a0[i];
            f[i + 4]  = (bf16_t)a1[i];
            f[i + 8]  = (bf16_t)a2[i];
            f[i + 12] = (bf16_t)a3[i];
        }
        qf[ki] = f;
    }

    // ---- scores = q @ Kp^T : 16 N-tiles of 16 kk's, Kdim = 64 ----
    v8f acc[16];
#pragma unroll
    for (int t = 0; t < 16; ++t) acc[t] = (v8f)(0.0f);
#pragma unroll
    for (int t = 0; t < 16; ++t) {
        const int kkrow = t * 16 + l15;
#pragma unroll
        for (int ki = 0; ki < 2; ++ki) {
            v16bf bf = frag2(&Kpb[kkrow * DH + ki * 32 + half * 8],
                             &Kpb[kkrow * DH + ki * 32 + 16 + half * 8]);
            acc[t] = wmma_bf16(qf[ki], bf, acc[t]);
        }
    }

    // ---- softmax over kk (256) per row, in registers ----
    const float scale = 0.125f;   // 1/sqrt(64)
#pragma unroll
    for (int j = 0; j < 8; ++j) {
        float mx = -3.0e38f;
#pragma unroll
        for (int t = 0; t < 16; ++t) {
            acc[t][j] *= scale;
            mx = fmaxf(mx, acc[t][j]);
        }
#pragma unroll
        for (int sh = 1; sh < 16; sh <<= 1) mx = fmaxf(mx, __shfl_xor(mx, sh, 32));
        float sm = 0.0f;
#pragma unroll
        for (int t = 0; t < 16; ++t) {
            float e = __expf(acc[t][j] - mx);
            acc[t][j] = e;
            sm += e;
        }
#pragma unroll
        for (int sh = 1; sh < 16; sh <<= 1) sm += __shfl_xor(sm, sh, 32);
        const float inv = 1.0f / sm;
#pragma unroll
        for (int t = 0; t < 16; ++t) acc[t][j] *= inv;
    }

    // ---- write probs to global output + stash bf16 copy in LDS ----
    float* pbase = probs_out + ((size_t)bh * SEQ + m0) * KP;
    bf16_t* Pw = &P[wave][0];
#pragma unroll
    for (int t = 0; t < 16; ++t)
#pragma unroll
        for (int j = 0; j < 8; ++j) {
            const int row = j + 8 * half;
            const int col = t * 16 + l15;
            pbase[(size_t)row * KP + col] = acc[t][j];
            Pw[row * KP + col] = (bf16_t)acc[t][j];
        }

    // ---- out = probs @ Vp : M=16, N=64 (4 tiles), Kdim=256 (8 iters) ----
    v8f oacc[4];
#pragma unroll
    for (int tn = 0; tn < 4; ++tn) oacc[tn] = (v8f)(0.0f);
#pragma unroll
    for (int ki = 0; ki < 8; ++ki) {
        v16bf af = frag2(&Pw[l15 * KP + ki * 32 + half * 8],
                         &Pw[l15 * KP + ki * 32 + 16 + half * 8]);
#pragma unroll
        for (int tn = 0; tn < 4; ++tn) {
            const int d = tn * 16 + l15;
            v16bf bf = frag2(&Vptb[d * KP + ki * 32 + half * 8],
                             &Vptb[d * KP + ki * 32 + 16 + half * 8]);
            oacc[tn] = wmma_bf16(af, bf, oacc[tn]);
        }
    }

    float* obase = out + ((size_t)bh * SEQ + m0) * DH;
#pragma unroll
    for (int tn = 0; tn < 4; ++tn)
#pragma unroll
        for (int j = 0; j < 8; ++j)
            obase[(size_t)(j + 8 * half) * DH + tn * 16 + l15] = oacc[tn][j];
}

// =====================================================================
extern "C" void kernel_launch(void* const* d_in, const int* in_sizes, int n_in,
                              void* d_out, int out_size, void* d_ws, size_t ws_size,
                              hipStream_t stream) {
    const float* q    = (const float*)d_in[0];
    const float* k    = (const float*)d_in[1];
    const float* v    = (const float*)d_in[2];
    const float* mask = (const float*)d_in[3];
    const float* pk   = (const float*)d_in[4];
    const float* pv   = (const float*)d_in[5];

    const size_t projElems = (size_t)BH * KP * DH;    // 2,097,152 floats each
    float* wsK = (float*)d_ws;
    float* wsV = wsK + projElems;

    // zero the split-K accumulation buffers (capture-safe)
    hipMemsetAsync(d_ws, 0, 2 * projElems * sizeof(float), stream);

    linf_proj_kernel<<<dim3(BH, 8), 256, 0, stream>>>(k, v, mask, pk, pv, wsK, wsV);

    float* out_o = (float*)d_out;                           // [B,H,S,D]
    float* out_p = out_o + (size_t)BH * SEQ * DH;           // [B,H,S,K]
    linf_attn_kernel<<<dim3(BH, SEQ / 128), 256, 0, stream>>>(q, wsK, wsV, out_o, out_p);
}
